// NaiveEuclideanGNN_36773509988630
// MI455X (gfx1250) — matrix operand order repaired and checked by reading, not compiled, not run on hardware
//
#include <hip/hip_runtime.h>
#include <hip/hip_bf16.h>

typedef __attribute__((ext_vector_type(16))) _Float16 v16h;
typedef __attribute__((ext_vector_type(8)))  _Float16 v8h;
typedef __attribute__((ext_vector_type(4)))  _Float16 v4h;
typedef __attribute__((ext_vector_type(8)))  float    v8f;
typedef __attribute__((ext_vector_type(4)))  int      v4i;
typedef __attribute__((address_space(1))) v4i* as1_v4i_p;  // global int4*
typedef __attribute__((address_space(3))) v4i* as3_v4i_p;  // LDS int4*

#define N_NODES  100000
#define N_EDGES  1600000
#define N_GRAPHS 1024
#define HID      128
#define ROWS_PB  64                      // rows per GEMM block
#define N_PAD    ((N_NODES + ROWS_PB - 1) / ROWS_PB * ROWS_PB)  // 100032

// ---------------------------------------------------------------- utilities
__global__ void cvt_f32_to_f16(const float* __restrict__ in,
                               _Float16* __restrict__ out, int n) {
  int i = blockIdx.x * blockDim.x + threadIdx.x;
  int stride = gridDim.x * blockDim.x;
  for (; i < n; i += stride) out[i] = (_Float16)in[i];
}

__global__ void zero_f32(float* __restrict__ p, int n) {
  int i = blockIdx.x * blockDim.x + threadIdx.x;
  int stride = gridDim.x * blockDim.x;
  for (; i < n; i += stride) p[i] = 0.0f;
}

// ----------------------------------------------------- node initialization
__global__ void node_init(const int* __restrict__ z,
                          const float* __restrict__ pos,
                          const float* __restrict__ emb,
                          const float* __restrict__ Wpos,
                          const float* __restrict__ bpos,
                          _Float16* __restrict__ A0, int N) {
  int i = blockIdx.x;
  int t = threadIdx.x;  // 128 threads: one feature each
  if (i >= N) return;
  int zi = z[i];
  float a  = emb[(size_t)zi * HID + t];
  float p0 = pos[(size_t)i * 3 + 0];
  float p1 = pos[(size_t)i * 3 + 1];
  float p2 = pos[(size_t)i * 3 + 2];
  float pe = bpos[t] + p0 * Wpos[t * 3 + 0] + p1 * Wpos[t * 3 + 1] +
             p2 * Wpos[t * 3 + 2];
  A0[(size_t)i * 256 + t]       = (_Float16)a;
  A0[(size_t)i * 256 + HID + t] = (_Float16)pe;
}

// ------------------------------------------------------------- WMMA GEMM
// out[N][128] = relu?( A[N][K] @ W[128][K]^T + bias )
// Block = 256 threads (8 waves), tile = 64 rows x 128 cols.
// Wave w owns column tile [16w, 16w+16) and 4 independent M-tile accumulators
// (B fragment reused 4x; 4 independent WMMA chains hide D->C latency).
// A tile staged into LDS via async global->LDS copy (ASYNCcnt path).
__global__ __launch_bounds__(256)
void gemm_wmma(const _Float16* __restrict__ A, const _Float16* __restrict__ W,
               const float* __restrict__ bias, _Float16* __restrict__ out,
               int N, int K, int relu) {
  __shared__ _Float16 ldsA[ROWS_PB * 256];  // 32 KB max (K<=256)
  const int tid  = threadIdx.x;
  const int row0 = blockIdx.x * ROWS_PB;

  // ---- stage the 64-row A tile into LDS (rows beyond N are padded junk) ----
  {
    const int nvec = (ROWS_PB * K) / 8;  // 16B vectors
    char* srcb = (char*)(A + (size_t)row0 * K);
    char* dstb = (char*)ldsA;
#if __has_builtin(__builtin_amdgcn_global_load_async_to_lds_b128)
    for (int v = tid; v < nvec; v += 256) {
      __builtin_amdgcn_global_load_async_to_lds_b128(
          (as1_v4i_p)(srcb + (size_t)v * 16),
          (as3_v4i_p)(dstb + (size_t)v * 16),
          0, 0);
    }
#if __has_builtin(__builtin_amdgcn_s_wait_asynccnt)
    __builtin_amdgcn_s_wait_asynccnt(0);
#else
    asm volatile("s_wait_asynccnt 0x0" ::: "memory");
#endif
#else
    const uint4* src = (const uint4*)srcb;
    uint4* dst = (uint4*)dstb;
    for (int v = tid; v < nvec; v += 256) dst[v] = src[v];
#endif
  }
  __syncthreads();

  const int lane = tid & 31;
  const int wave = tid >> 5;
  const int m = lane & 15;     // row within M-tile / col within N-tile
  const int g = lane >> 4;     // lane half-group
  const int n = wave * 16 + m; // output column

  v8f acc[4] = {{}, {}, {}, {}};
  const _Float16* wrow = W + (size_t)n * K;

  for (int k0 = 0; k0 < K; k0 += 32) {
    // B fragment: lane holds col n, elems 0-7 -> K = k0+8g+i ; elems 8-15 -> +16
    v8h blo = *(const v8h*)(wrow + k0 + g * 8);
    v8h bhi = *(const v8h*)(wrow + k0 + g * 8 + 16);
    v16h b;
#pragma unroll
    for (int i = 0; i < 8; ++i) { b[i] = blo[i]; b[i + 8] = bhi[i]; }
#pragma unroll
    for (int mt = 0; mt < 4; ++mt) {
      // A fragment: elems 0-7 -> K = k0+8g+i ; elems 8-15 -> +16
      const _Float16* arow = ldsA + (size_t)(mt * 16 + m) * K + k0 + g * 8;
      v8h lo = *(const v8h*)(arow);
      v8h hi = *(const v8h*)(arow + 16);
      v16h a;
#pragma unroll
      for (int i = 0; i < 8; ++i) { a[i] = lo[i]; a[i + 8] = hi[i]; }
      acc[mt] = __builtin_amdgcn_wmma_f32_16x16x32_f16(
          false, a, false, b, (short)0, acc[mt], false, false);
    }
  }

  const float bn = bias[n];
#pragma unroll
  for (int mt = 0; mt < 4; ++mt) {
#pragma unroll
    for (int r = 0; r < 8; ++r) {  // C layout: VGPR r -> M = r + 8g, N = m
      int row = row0 + mt * 16 + r + 8 * g;
      if (row < N) {
        float v = acc[mt][r] + bn;
        if (relu) v = fmaxf(v, 0.0f);
        out[(size_t)row * HID + n] = (_Float16)v;
      }
    }
  }
}

// ------------------------------------------------------------ edge scatter
// aggr[dst[e]][:] += x[src[e]][:]   (one wave per edge; 4 feats per lane)
__global__ void edge_scatter(const int* __restrict__ src,
                             const int* __restrict__ dst,
                             const _Float16* __restrict__ x,
                             float* __restrict__ aggr, int E) {
  int tid  = blockIdx.x * blockDim.x + threadIdx.x;
  int e    = tid >> 5;
  int lane = tid & 31;
  if (e >= E) return;
  int s = src[e];
  int d = dst[e];
  v4h v = *(const v4h*)(x + (size_t)s * HID + lane * 4);
  float* ad = aggr + (size_t)d * HID + lane * 4;
  atomicAdd(ad + 0, (float)v[0]);
  atomicAdd(ad + 1, (float)v[1]);
  atomicAdd(ad + 2, (float)v[2]);
  atomicAdd(ad + 3, (float)v[3]);
}

// hin = (f16)(aggr + x)
__global__ void fuse_add(const float* __restrict__ aggr,
                         const _Float16* __restrict__ x,
                         _Float16* __restrict__ hin, int n) {
  int i = blockIdx.x * blockDim.x + threadIdx.x;
  int stride = gridDim.x * blockDim.x;
  for (; i < n; i += stride) hin[i] = (_Float16)(aggr[i] + (float)x[i]);
}

// ------------------------------------------------------------------ pooling
__global__ void pool_scatter(const int* __restrict__ batch,
                             const _Float16* __restrict__ x,
                             float* __restrict__ gout, int N) {
  int i = blockIdx.x;
  int t = threadIdx.x;  // 128 threads
  if (i >= N) return;
  int b = batch[i];
  atomicAdd(&gout[(size_t)b * HID + t], (float)x[(size_t)i * HID + t]);
}

// ---------------------------------------------------------------- MLP head
__global__ void head_mlp(const float* __restrict__ gfeat,
                         const float* __restrict__ Wp1,
                         const float* __restrict__ bp1,
                         const float* __restrict__ Wp2,
                         const float* __restrict__ bp2,
                         float* __restrict__ out) {
  __shared__ float sm[HID];
  int gidx = blockIdx.x;
  int t = threadIdx.x;  // 128 threads
  const float* gr = gfeat + (size_t)gidx * HID;
  float acc = bp1[t];
  const float* wr = Wp1 + (size_t)t * HID;
  for (int k = 0; k < HID; ++k) acc += gr[k] * wr[k];
  float h = fmaxf(acc, 0.0f);
  sm[t] = h * Wp2[t];
  __syncthreads();
  for (int s = 64; s > 0; s >>= 1) {
    if (t < s) sm[t] += sm[t + s];
    __syncthreads();
  }
  if (t == 0) out[gidx] = sm[0] + bp2[0];
}

// ================================================================== launch
extern "C" void kernel_launch(void* const* d_in, const int* in_sizes, int n_in,
                              void* d_out, int out_size, void* d_ws,
                              size_t ws_size, hipStream_t stream) {
  const int*   z       = (const int*)d_in[0];
  const float* pos     = (const float*)d_in[1];
  const int*   edge    = (const int*)d_in[2];   // [2][E]: src then dst
  const int*   batch   = (const int*)d_in[3];
  const float* emb     = (const float*)d_in[4];
  const float* W_pos   = (const float*)d_in[5];
  const float* b_pos   = (const float*)d_in[6];
  const float* W_comb  = (const float*)d_in[7];
  const float* b_comb  = (const float*)d_in[8];
  const float* gW1     = (const float*)d_in[9];
  const float* gb1     = (const float*)d_in[10];
  const float* gW2     = (const float*)d_in[11];
  const float* gb2     = (const float*)d_in[12];
  const float* W_p1    = (const float*)d_in[13];
  const float* b_p1    = (const float*)d_in[14];
  const float* W_p2    = (const float*)d_in[15];
  const float* b_p2    = (const float*)d_in[16];
  float* out = (float*)d_out;

  // ---------------- workspace layout (all 256B aligned) ----------------
  char* ws = (char*)d_ws;
  size_t off = 0;
  auto alloc = [&](size_t bytes) -> void* {
    void* p = ws + off;
    off = (off + bytes + 255) & ~(size_t)255;
    return p;
  };
  // f16 weights: Wc[128][256], gW1[3][128][128], gW2[3][128][128]
  _Float16* wts   = (_Float16*)alloc((32768 + 49152 + 49152) * 2);
  _Float16* x16   = (_Float16*)alloc((size_t)N_PAD * HID * 2);
  _Float16* h16   = (_Float16*)alloc((size_t)N_PAD * HID * 2);
  _Float16* hin16 = (_Float16*)alloc((size_t)N_PAD * HID * 2);
  float*    aggr  = (float*)alloc((size_t)N_PAD * HID * 4);
  _Float16* A0    = (_Float16*)aggr;  // alias: A0[N_PAD][256] f16, same bytes
  float*    gfeat = (float*)alloc((size_t)N_GRAPHS * HID * 4);
  (void)ws_size; (void)in_sizes; (void)n_in; (void)out_size;

  _Float16* wc  = wts;
  _Float16* wg1 = wts + 32768;
  _Float16* wg2 = wts + 32768 + 49152;

  // ---------------- weight conversion (f32 -> f16) ----------------
  cvt_f32_to_f16<<<128, 256, 0, stream>>>(W_comb, wc, 32768);
  cvt_f32_to_f16<<<192, 256, 0, stream>>>(gW1, wg1, 49152);
  cvt_f32_to_f16<<<192, 256, 0, stream>>>(gW2, wg2, 49152);

  // ---------------- node embedding + pos linear ----------------
  node_init<<<N_NODES, HID, 0, stream>>>(z, pos, emb, W_pos, b_pos, A0, N_NODES);

  const int gemm_blocks = N_PAD / ROWS_PB;  // 1563

  // x = relu(A0 @ Wc.T + b_comb)     [K = 256]
  gemm_wmma<<<gemm_blocks, 256, 0, stream>>>(A0, wc, b_comb, x16,
                                             N_NODES, 256, 1);

  const int* esrc = edge;
  const int* edst = edge + N_EDGES;
  const int feat_elems = N_NODES * HID;

  for (int l = 0; l < 3; ++l) {
    zero_f32<<<4096, 256, 0, stream>>>(aggr, feat_elems);
    edge_scatter<<<(N_EDGES * 32) / 256, 256, 0, stream>>>(esrc, edst, x16,
                                                           aggr, N_EDGES);
    fuse_add<<<4096, 256, 0, stream>>>(aggr, x16, hin16, feat_elems);
    // h = relu(hin @ W1.T + b1)
    gemm_wmma<<<gemm_blocks, 256, 0, stream>>>(
        hin16, wg1 + (size_t)l * 16384, gb1 + l * HID, h16, N_NODES, HID, 1);
    // x = (h @ W2.T + b2), relu except last layer
    gemm_wmma<<<gemm_blocks, 256, 0, stream>>>(
        h16, wg2 + (size_t)l * 16384, gb2 + l * HID, x16, N_NODES, HID,
        (l < 2) ? 1 : 0);
  }

  // ---------------- pooling + head ----------------
  zero_f32<<<64, 256, 0, stream>>>(gfeat, N_GRAPHS * HID);
  pool_scatter<<<N_NODES, HID, 0, stream>>>(batch, x16, gfeat, N_NODES);
  head_mlp<<<N_GRAPHS, HID, 0, stream>>>(gfeat, W_p1, b_p1, W_p2, b_p2, out);
}
